// sLSTM_19198503813779
// MI455X (gfx1250) — matrix operand order repaired and checked by reading, not compiled, not run on hardware
//
#include <hip/hip_runtime.h>
#include <math.h>

// ---------------- problem constants ----------------
constexpr int NB    = 8192;   // batch
constexpr int NI    = 2048;   // INP
constexpr int NH    = 8;      // heads
constexpr int NDH   = 256;    // head dim
constexpr int NHD   = 2048;   // H*DH
constexpr int NPROJ = 2730;   // int(4/3 * 2048)
constexpr float EPSV = 1e-5f;

// ---------------- vector types ----------------
typedef __attribute__((ext_vector_type(16))) __bf16        v16bf;
typedef __attribute__((ext_vector_type(8)))  float         v8f;
typedef __attribute__((ext_vector_type(4)))  unsigned int  u32x4;
typedef __attribute__((ext_vector_type(4)))  int           i32x4;
typedef __attribute__((ext_vector_type(8)))  int           i32x8;

#if defined(__has_builtin)
#  if __has_builtin(__builtin_amdgcn_tensor_load_to_lds) && \
      __has_builtin(__builtin_amdgcn_s_wait_tensorcnt)
#    define HAVE_TDM 1
#  else
#    define HAVE_TDM 0
#  endif
#else
#  define HAVE_TDM 0
#endif

__device__ inline v8f wmma_bf16(v16bf a, v16bf b, v8f c) {
    return __builtin_amdgcn_wmma_f32_16x16x32_bf16(false, a, false, b,
                                                   (short)0, c, false, false);
}

__device__ inline __bf16 f2bf(float f) {
    unsigned u; __builtin_memcpy(&u, &f, 4);
    unsigned r = u + 0x7FFFu + ((u >> 16) & 1u);   // round-to-nearest-even
    unsigned short h = (unsigned short)(r >> 16);
    __bf16 o; __builtin_memcpy(&o, &h, 2);
    return o;
}
__device__ inline unsigned short ld_bf_bits(const __bf16* p) {
    unsigned short h; __builtin_memcpy(&h, p, 2); return h;
}
__device__ inline void st_bf_bits(__bf16* p, unsigned short h) {
    __builtin_memcpy(p, &h, 2);
}
__device__ inline float gelu_exact(float x) {
    return 0.5f * x * (1.0f + erff(x * 0.70710678118654752f));
}

// ---------------- LDS layout ----------------
// A tile: 128 rows x 32 k, row stride 48 elems (96 B, 16B-aligned rows).
// 96 B = 64 B data + 32 B pad == TDM pad_interval=16 DW, pad_amount=8 DW.
// B tile: staged TRANSPOSED (n-major): 128 n-rows x 32 k, stride 48 elems.
constexpr int SA  = 48;
constexpr int SBT = 48;
constexpr int TILE_ELEMS = 128 * 48;   // 12 KB per buffer

// ---- TDM: one DMA per workgroup stages the 128x32 bf16 A tile ----
__device__ inline void stage_A(const __bf16* __restrict__ gsrc, int ld,
                               __bf16* Al, int validCols, int tid) {
#if HAVE_TDM
    if ((tid >> 5) == 0) {                       // wave 0 issues the descriptor
        unsigned long long ga = (unsigned long long)(const void*)gsrc;
        unsigned lds = (unsigned)(unsigned long long)(void*)Al;
        u32x4 g0;
        g0[0] = 1u;                                 // count=1, user mode
        g0[1] = lds;                                // lds_addr
        g0[2] = (unsigned)ga;                       // global_addr[31:0]
        g0[3] = (unsigned)(ga >> 32) | 0x80000000u; // addr[56:32] | type=2
        i32x8 g1;
        g1[0] = (int)((1u << 16)                    // data_size = 2B
                    | (1u << 20)                    // pad_enable
                    | (3u << 22)                    // pad_interval: 16 DWORDs (row)
                    | (7u << 25));                  // pad_amount: 8 DWORDs (32 B)
        g1[1] = (int)(((unsigned)validCols & 0xFFFFu) << 16);       // tensor_dim0 lo
        g1[2] = (int)(((unsigned)validCols >> 16) | (128u << 16));  // hi | tensor_dim1 lo
        g1[3] = (int)(32u << 16);                   // tensor_dim1 hi | tile_dim0=32
        g1[4] = 128;                                // tile_dim1=128, tile_dim2=0
        g1[5] = ld;                                 // tensor_dim0_stride (elements)
        g1[6] = 0; g1[7] = 0;
        i32x4 z4 = {0, 0, 0, 0};
#if __clang_major__ >= 23
        i32x8 z8 = {0, 0, 0, 0, 0, 0, 0, 0};
        __builtin_amdgcn_tensor_load_to_lds(g0, g1, z4, z4, z8, 0);
#else
        __builtin_amdgcn_tensor_load_to_lds(g0, g1, z4, z4, 0);
#endif
    }
#else
    #pragma unroll
    for (int it = 0; it < 2; ++it) {
        int c = tid + it * 256;
        int r = c >> 2;
        int cc = (c & 3) * 8;
        const __bf16* src = gsrc + (size_t)r * ld + cc;
        __bf16* dst = Al + r * SA + cc;
        if (cc + 8 <= validCols) {
            const unsigned int* s32 = (const unsigned int*)src;
            unsigned int* d32 = (unsigned int*)dst;
            d32[0] = s32[0]; d32[1] = s32[1]; d32[2] = s32[2]; d32[3] = s32[3];
        } else {
            #pragma unroll
            for (int j = 0; j < 8; ++j)
                st_bf_bits(dst + j,
                           (cc + j < validCols) ? ld_bf_bits(src + j) : (unsigned short)0);
        }
    }
#endif
}
__device__ inline void stage_A_wait(int tid) {
#if HAVE_TDM
    if ((tid >> 5) == 0) __builtin_amdgcn_s_wait_tensorcnt((short)0);
#endif
}

// ---- stage B tile transposed into LDS (k-major global -> n-major LDS) ----
// GUARD=false compiles the pure fast path (no exec-mask trees in hot loops).
template<bool GUARD>
__device__ inline void stage_Bt(const __bf16* __restrict__ g, int ld,
                                int k0, int n0, __bf16* Bt, int tid,
                                int kLim = 0x40000000, int nLim = 0x40000000) {
    #pragma unroll
    for (int it = 0; it < 2; ++it) {
        int c = tid + it * 256;
        int r  = c >> 4;            // k row within tile (0..31)
        int cc = (c & 15) * 8;      // n offset (0..120)
        const __bf16* src = g + (size_t)(k0 + r) * ld + (n0 + cc);
        unsigned short e[8];
        if (!GUARD || ((k0 + r) < kLim && (n0 + cc + 8) <= nLim)) {
            const unsigned int* s32 = (const unsigned int*)src;  // 4B-aligned always
            #pragma unroll
            for (int j = 0; j < 4; ++j) {
                unsigned t = s32[j];
                e[2 * j]     = (unsigned short)t;
                e[2 * j + 1] = (unsigned short)(t >> 16);
            }
        } else {
            #pragma unroll
            for (int j = 0; j < 8; ++j)
                e[j] = ((k0 + r) < kLim && (n0 + cc + j) < nLim)
                           ? ld_bf_bits(src + j) : (unsigned short)0;
        }
        #pragma unroll
        for (int j = 0; j < 8; ++j) st_bf_bits(Bt + (cc + j) * SBT + r, e[j]);
    }
}

// ---- fragment loads: two 16B vector LDS reads each (ds_load_b128) ----
union FragU { u32x4 q[2]; v16bf v; };

__device__ inline v16bf frag_a(const __bf16* Al, int mBase, int lane) {
    const int half = lane >> 4, r = lane & 15;
    const u32x4* row = (const u32x4*)(Al + (mBase + r) * SA);  // 96B rows, 16B aligned
    FragU f;
    f.q[0] = row[half];        // k = half*8 .. half*8+7
    f.q[1] = row[2 + half];    // k = 16+half*8 .. 16+half*8+7
    return f.v;
}
__device__ inline v16bf frag_b(const __bf16* Bt, int nBase, int lane) {
    const int half = lane >> 4, r = lane & 15;
    const u32x4* p = (const u32x4*)(Bt + (nBase + r) * SBT);
    FragU f;
    f.q[0] = p[half];          // k = half*8 .. half*8+7
    f.q[1] = p[2 + half];      // k = 16+half*8 .. 16+half*8+7
    return f.v;
}

// ---- one K-step of WMMA on a 128x128 tile (this wave: 32x64 = 2x4 frags) ----
__device__ inline void mma_step(const __bf16* Ac, const __bf16* Bc,
                                int mB, int nB, int lane, v8f (&acc)[2][4]) {
    v16bf a0 = frag_a(Ac, mB, lane);
    v16bf a1 = frag_a(Ac, mB + 16, lane);
    #pragma unroll
    for (int ni = 0; ni < 4; ++ni) {
        v16bf b = frag_b(Bc, nB + 16 * ni, lane);
        acc[0][ni] = wmma_bf16(a0, b, acc[0][ni]);
        acc[1][ni] = wmma_bf16(a1, b, acc[1][ni]);
    }
}

// ---------------- kernel 0: fp32 -> bf16 convert ----------------
__global__ __launch_bounds__(256) void f32_to_bf16(const float* __restrict__ s,
                                                   __bf16* __restrict__ d, int n) {
    for (int i = blockIdx.x * 256 + threadIdx.x; i < n; i += gridDim.x * 256)
        d[i] = f2bf(s[i]);
}

// ---------------- kernel 1: LayerNorm -> bf16 ----------------
__global__ __launch_bounds__(256) void layernorm_k(const float* __restrict__ X,
                                                   const float* __restrict__ w,
                                                   const float* __restrict__ b,
                                                   __bf16* __restrict__ Xb) {
    __shared__ float red[16];
    const int row = blockIdx.x, tid = threadIdx.x;
    const float* x = X + (size_t)row * NI;
    float v[8], s = 0.f, s2 = 0.f;
    #pragma unroll
    for (int j = 0; j < 8; ++j) { v[j] = x[tid + 256 * j]; s += v[j]; s2 += v[j] * v[j]; }
    #pragma unroll
    for (int o = 16; o > 0; o >>= 1) { s += __shfl_xor(s, o, 32); s2 += __shfl_xor(s2, o, 32); }
    if ((tid & 31) == 0) { red[tid >> 5] = s; red[8 + (tid >> 5)] = s2; }
    __syncthreads();
    float ts = 0.f, ts2 = 0.f;
    #pragma unroll
    for (int k2 = 0; k2 < 8; ++k2) { ts += red[k2]; ts2 += red[8 + k2]; }
    const float mu  = ts / NI;
    const float inv = rsqrtf(ts2 / NI - mu * mu + EPSV);
    #pragma unroll
    for (int j = 0; j < 8; ++j) {
        int c = tid + 256 * j;
        Xb[(size_t)row * NI + c] = f2bf((v[j] - mu) * inv * w[c] + b[c]);
    }
}

// ---------------- kernel 2: gate GEMM  pre[g] = x@W_g + b_g + h@R_g ------
// Double-buffered: TDM stages A(k+1) and waves transpose-stage B(k+1) while
// WMMAs consume tile k. One barrier per K-step. Phase-1 tail prestages the
// first recurrent (R) tile so the pipeline never drains.
__global__ __launch_bounds__(256) void gate_gemm(
    const __bf16* __restrict__ X, const __bf16* __restrict__ Hb,
    const __bf16* __restrict__ W4, const __bf16* __restrict__ R4,
    const float* __restrict__ bi, const float* __restrict__ bff,
    const float* __restrict__ bz, const float* __restrict__ bo,
    float* __restrict__ pre) {
    __shared__ __bf16 Al0[TILE_ELEMS], Al1[TILE_ELEMS];
    __shared__ __bf16 Bt0[TILE_ELEMS], Bt1[TILE_ELEMS];
    const int g  = blockIdx.z;
    const int n0 = blockIdx.x * 128;
    const int m0 = blockIdx.y * 128;
    const int head = n0 / NDH;
    const __bf16* W = W4 + (size_t)g * NI * NHD;
    const __bf16* R = R4 + ((size_t)g * NH + head) * NDH * NDH;
    const float* bias = (g == 0) ? bi : (g == 1) ? bff : (g == 2) ? bz : bo;

    const int tid = threadIdx.x, lane = tid & 31, wid = tid >> 5;
    const int mB = (wid & 3) * 32, nB = (wid >> 2) * 64;

    __bf16 *Ac = Al0, *An = Al1, *Bc = Bt0, *Bn = Bt1;
    v8f acc[2][4] = {};

    // prologue: stage first x/W tile
    stage_A(X + (size_t)m0 * NI, NI, Ac, 32, tid);
    stage_Bt<false>(W, NHD, 0, n0, Bc, tid);
    stage_A_wait(tid);
    __syncthreads();

    // phase 1: x @ W_g (K = 2048)
    for (int kb = 0; kb < NI; kb += 32) {
        if (kb + 32 < NI) {
            stage_A(X + (size_t)m0 * NI + kb + 32, NI, An, 32, tid);
            stage_Bt<false>(W, NHD, kb + 32, n0, Bn, tid);
            __builtin_prefetch(W + (size_t)(kb + 64) * NHD + n0 + (lane << 2), 0, 1);
        } else {   // prestage first recurrent tile
            stage_A(Hb + (size_t)m0 * NHD + head * NDH, NHD, An, 32, tid);
            stage_Bt<false>(R, NDH, 0, n0 - head * NDH, Bn, tid);
        }
        mma_step(Ac, Bc, mB, nB, lane, acc);
        stage_A_wait(tid);
        __syncthreads();
        __bf16* t = Ac; Ac = An; An = t;
        t = Bc; Bc = Bn; Bn = t;
    }
    // phase 2: block-diagonal h[:, head*256:+256] @ R_g[head] (K = 256)
    for (int kb = 0; kb < NDH; kb += 32) {
        bool more = kb + 32 < NDH;
        if (more) {
            stage_A(Hb + (size_t)m0 * NHD + head * NDH + kb + 32, NHD, An, 32, tid);
            stage_Bt<false>(R, NDH, kb + 32, n0 - head * NDH, Bn, tid);
        }
        mma_step(Ac, Bc, mB, nB, lane, acc);
        if (more) stage_A_wait(tid);
        __syncthreads();
        __bf16* t = Ac; Ac = An; An = t;
        t = Bc; Bc = Bn; Bn = t;
    }
    const int half = lane >> 4, r = lane & 15;
    #pragma unroll
    for (int mi = 0; mi < 2; ++mi)
        #pragma unroll
        for (int ni = 0; ni < 4; ++ni)
            #pragma unroll
            for (int v = 0; v < 8; ++v) {
                int m = m0 + mB + mi * 16 + v + 8 * half;
                int n = n0 + nB + ni * 16 + r;
                pre[(size_t)g * NB * NHD + (size_t)m * NHD + n] = acc[mi][ni][v] + bias[n];
            }
}

// ---------------- kernel 3: sLSTM pointwise state update ----------------
__global__ __launch_bounds__(256) void slstm_pointwise(
    const float* __restrict__ pre, const float* __restrict__ cp,
    const float* __restrict__ np, const float* __restrict__ mp,
    float* __restrict__ co, float* __restrict__ no,
    float* __restrict__ ho, float* __restrict__ mo) {
    const size_t S = (size_t)NB * NHD;
    size_t i = (size_t)blockIdx.x * 256 + threadIdx.x;
    float it = pre[i], ft = pre[S + i], zt = pre[2 * S + i], ot = pre[3 * S + i];
    float m_p = mp[i];
    float m_t = fmaxf(ft + m_p, it);
    float i_g = expf(it - m_t);
    float f_g = expf(ft - m_t + m_p);
    float z_g = tanhf(zt);
    float o_g = 1.f / (1.f + expf(-ot));
    float c_t = f_g * cp[i] + i_g * z_g;
    float n_t = f_g * np[i] + i_g;
    float h_t = o_g * (c_t / n_t);
    co[i] = c_t; no[i] = n_t; ho[i] = h_t; mo[i] = m_t;
}

// ---------------- kernel 4: GroupNorm (per row, per head) -> bf16 --------
__global__ __launch_bounds__(256) void groupnorm_k(const float* __restrict__ Hh,
                                                   const float* __restrict__ gw,
                                                   const float* __restrict__ gb,
                                                   __bf16* __restrict__ Ob) {
    const int row = blockIdx.x, tid = threadIdx.x;
    const int head = tid >> 5, lane = tid & 31;
    const float* x = Hh + (size_t)row * NHD + head * NDH;
    float v[8], s = 0.f, s2 = 0.f;
    #pragma unroll
    for (int j = 0; j < 8; ++j) { v[j] = x[lane + 32 * j]; s += v[j]; s2 += v[j] * v[j]; }
    #pragma unroll
    for (int o = 16; o > 0; o >>= 1) { s += __shfl_xor(s, o, 32); s2 += __shfl_xor(s2, o, 32); }
    const float mu  = s / NDH;
    const float inv = rsqrtf(s2 / NDH - mu * mu + EPSV);
    #pragma unroll
    for (int j = 0; j < 8; ++j) {
        int c = head * NDH + lane + 32 * j;
        Ob[(size_t)row * NHD + c] = f2bf((v[j] - mu) * inv * gw[c] + gb[c]);
    }
}

// ---- kernel 5: up GEMM, dual-column fused: proj = up1 + gelu(up2) -------
// Body templated on EDGE so only the single rightmost N-tile compiles guards.
template<bool EDGE>
__device__ inline void up_body(const __bf16* __restrict__ O,
                               const __bf16* __restrict__ Wu,
                               const float* __restrict__ bup,
                               __bf16* __restrict__ Pj,
                               int m0, int n0, __bf16* Al, __bf16* B1t, __bf16* B2t,
                               int tid, int lane, int mB, int nB) {
    v8f acc1[2][4] = {}, acc2[2][4] = {};
    for (int kb = 0; kb < NHD; kb += 32) {
        stage_A(O + (size_t)m0 * NHD + kb, NHD, Al, 32, tid);
        stage_Bt<EDGE>(Wu, 2 * NPROJ, kb, n0, B1t, tid, NHD, NPROJ);
        stage_Bt<EDGE>(Wu, 2 * NPROJ, kb, n0 + NPROJ, B2t, tid, NHD, 2 * NPROJ);
        stage_A_wait(tid);
        __syncthreads();
        v16bf a0 = frag_a(Al, mB, lane);
        v16bf a1 = frag_a(Al, mB + 16, lane);
        #pragma unroll
        for (int ni = 0; ni < 4; ++ni) {
            v16bf b1 = frag_b(B1t, nB + 16 * ni, lane);
            acc1[0][ni] = wmma_bf16(a0, b1, acc1[0][ni]);
            acc1[1][ni] = wmma_bf16(a1, b1, acc1[1][ni]);
            v16bf b2 = frag_b(B2t, nB + 16 * ni, lane);
            acc2[0][ni] = wmma_bf16(a0, b2, acc2[0][ni]);
            acc2[1][ni] = wmma_bf16(a1, b2, acc2[1][ni]);
        }
        __syncthreads();
    }
    const int half = lane >> 4, r = lane & 15;
    #pragma unroll
    for (int mi = 0; mi < 2; ++mi)
        #pragma unroll
        for (int ni = 0; ni < 4; ++ni)
            #pragma unroll
            for (int v = 0; v < 8; ++v) {
                int n = n0 + nB + ni * 16 + r;
                if (!EDGE || n < NPROJ) {
                    int m = m0 + mB + mi * 16 + v + 8 * half;
                    float u1 = acc1[mi][ni][v] + bup[n];
                    float u2 = acc2[mi][ni][v] + bup[n + NPROJ];
                    Pj[(size_t)m * NPROJ + n] = f2bf(u1 + gelu_exact(u2));
                }
            }
}

__global__ __launch_bounds__(256) void up_gemm(
    const __bf16* __restrict__ O, const __bf16* __restrict__ Wu,
    const float* __restrict__ bup, __bf16* __restrict__ Pj) {
    __shared__ __bf16 Al[TILE_ELEMS];
    __shared__ __bf16 B1t[TILE_ELEMS];
    __shared__ __bf16 B2t[TILE_ELEMS];
    const int n0 = blockIdx.x * 128, m0 = blockIdx.y * 128;
    const int tid = threadIdx.x, lane = tid & 31, wid = tid >> 5;
    const int mB = (wid & 3) * 32, nB = (wid >> 2) * 64;
    if (n0 + 128 <= NPROJ)
        up_body<false>(O, Wu, bup, Pj, m0, n0, Al, B1t, B2t, tid, lane, mB, nB);
    else
        up_body<true>(O, Wu, bup, Pj, m0, n0, Al, B1t, B2t, tid, lane, mB, nB);
}

// ---- kernel 6: down GEMM + bias + residual: out = proj@Wd + b_dn + seq --
// Double-buffered; only the staging of the final K-tile (10 valid k) uses
// guards -- the A-tile edge is zero-filled by the TDM descriptor for free.
__global__ __launch_bounds__(256) void dn_gemm(
    const __bf16* __restrict__ P, const __bf16* __restrict__ Wd,
    const float* __restrict__ bdn, const float* __restrict__ seq,
    float* __restrict__ out) {
    __shared__ __bf16 Al0[TILE_ELEMS], Al1[TILE_ELEMS];
    __shared__ __bf16 Bt0[TILE_ELEMS], Bt1[TILE_ELEMS];
    const int n0 = blockIdx.x * 128, m0 = blockIdx.y * 128;
    const int tid = threadIdx.x, lane = tid & 31, wid = tid >> 5;
    const int mB = (wid & 3) * 32, nB = (wid >> 2) * 64;

    __bf16 *Ac = Al0, *An = Al1, *Bc = Bt0, *Bn = Bt1;
    v8f acc[2][4] = {};

    stage_A(P + (size_t)m0 * NPROJ, NPROJ, Ac, 32, tid);
    stage_Bt<false>(Wd, NI, 0, n0, Bc, tid);
    stage_A_wait(tid);
    __syncthreads();

    for (int kb = 0; kb < NPROJ; kb += 32) {       // K = 2730; last tile k=2720..2729
        const int nk = kb + 32;
        bool more = nk < NPROJ;
        if (more) {
            if (nk + 32 <= NPROJ) {                // full tile: fast path
                stage_A(P + (size_t)m0 * NPROJ + nk, NPROJ, An, 32, tid);
                stage_Bt<false>(Wd, NI, nk, n0, Bn, tid);
            } else {                               // edge tile: guarded staging
                stage_A(P + (size_t)m0 * NPROJ + nk, NPROJ, An, NPROJ - nk, tid);
                stage_Bt<true>(Wd, NI, nk, n0, Bn, tid, NPROJ, NI);
            }
            __builtin_prefetch(Wd + (size_t)(nk + 32) * NI + n0 + (lane << 2), 0, 1);
        }
        mma_step(Ac, Bc, mB, nB, lane, acc);
        if (more) stage_A_wait(tid);
        __syncthreads();
        __bf16* t = Ac; Ac = An; An = t;
        t = Bc; Bc = Bn; Bn = t;
    }
    const int half = lane >> 4, r = lane & 15;
    #pragma unroll
    for (int mi = 0; mi < 2; ++mi)
        #pragma unroll
        for (int ni = 0; ni < 4; ++ni)
            #pragma unroll
            for (int v = 0; v < 8; ++v) {
                int m = m0 + mB + mi * 16 + v + 8 * half;
                int n = n0 + nB + ni * 16 + r;
                out[(size_t)m * NI + n] = acc[mi][ni][v] + bdn[n] + seq[(size_t)m * NI + n];
            }
}

// ---------------- host launcher ----------------
extern "C" void kernel_launch(void* const* d_in, const int* in_sizes, int n_in,
                              void* d_out, int out_size, void* d_ws, size_t ws_size,
                              hipStream_t stream) {
    (void)in_sizes; (void)n_in; (void)out_size; (void)ws_size;
    const float* seq   = (const float*)d_in[0];
    const float* c_tm1 = (const float*)d_in[1];
    const float* n_tm1 = (const float*)d_in[2];
    const float* h_tm1 = (const float*)d_in[3];
    const float* m_tm1 = (const float*)d_in[4];
    const float* ln_w  = (const float*)d_in[5];
    const float* ln_b  = (const float*)d_in[6];
    const float* gn_w  = (const float*)d_in[7];
    const float* gn_b  = (const float*)d_in[8];
    const float* Ws[4] = { (const float*)d_in[9],  (const float*)d_in[11],
                           (const float*)d_in[13], (const float*)d_in[15] };
    const float* bs[4] = { (const float*)d_in[10], (const float*)d_in[12],
                           (const float*)d_in[14], (const float*)d_in[16] };
    const float* Rs[4] = { (const float*)d_in[17], (const float*)d_in[18],
                           (const float*)d_in[19], (const float*)d_in[20] };
    const float* W_up  = (const float*)d_in[21];
    const float* b_up  = (const float*)d_in[22];
    const float* W_dn  = (const float*)d_in[23];
    const float* b_dn  = (const float*)d_in[24];

    const size_t S = (size_t)NB * NHD;
    float* out = (float*)d_out;                 // [B, INP]
    float* c_o = out + (size_t)NB * NI;         // state tuple, flat in return order
    float* n_o = c_o + S;
    float* h_o = n_o + S;
    float* m_o = h_o + S;

    // workspace carve-out (256B aligned)
    char* w = (char*)d_ws;
    auto take = [&](size_t bytes) -> char* {
        char* p = w; w += (bytes + 255) & ~(size_t)255; return p;
    };
    float*  pre    = (float*) take(4 * S * sizeof(float));
    __bf16* xbf    = (__bf16*)take((size_t)NB * NI * 2);
    __bf16* hbf    = (__bf16*)take(S * 2);
    __bf16* Wbf    = (__bf16*)take((size_t)4 * NI * NHD * 2);
    __bf16* Rbf    = (__bf16*)take((size_t)4 * NH * NDH * NDH * 2);
    __bf16* Wupbf  = (__bf16*)take((size_t)NHD * 2 * NPROJ * 2);
    __bf16* Wdnbf  = (__bf16*)take((size_t)NPROJ * NI * 2);
    __bf16* obf    = (__bf16*)take(S * 2);
    __bf16* projbf = (__bf16*)take((size_t)NB * NPROJ * 2);

    layernorm_k<<<NB, 256, 0, stream>>>(seq, ln_w, ln_b, xbf);
    f32_to_bf16<<<2048, 256, 0, stream>>>(h_tm1, hbf, NB * NHD);
    for (int g = 0; g < 4; ++g) {
        f32_to_bf16<<<2048, 256, 0, stream>>>(Ws[g], Wbf + (size_t)g * NI * NHD, NI * NHD);
        f32_to_bf16<<<1024, 256, 0, stream>>>(Rs[g], Rbf + (size_t)g * NH * NDH * NDH,
                                              NH * NDH * NDH);
    }
    f32_to_bf16<<<2048, 256, 0, stream>>>(W_up, Wupbf, NHD * 2 * NPROJ);
    f32_to_bf16<<<2048, 256, 0, stream>>>(W_dn, Wdnbf, NPROJ * NI);

    gate_gemm<<<dim3(NHD / 128, NB / 128, 4), 256, 0, stream>>>(
        xbf, hbf, Wbf, Rbf, bs[0], bs[1], bs[2], bs[3], pre);

    slstm_pointwise<<<(unsigned)(S / 256), 256, 0, stream>>>(
        pre, c_tm1, n_tm1, m_tm1, c_o, n_o, h_o, m_o);

    groupnorm_k<<<NB, 256, 0, stream>>>(h_o, gn_w, gn_b, obf);

    up_gemm<<<dim3((NPROJ + 127) / 128, NB / 128), 256, 0, stream>>>(obf, Wupbf, b_up, projbf);

    dn_gemm<<<dim3(NI / 128, NB / 128), 256, 0, stream>>>(projbf, Wdnbf, b_dn, seq, out);
}
